// NHP_5806795784444
// MI455X (gfx1250) — matrix operand order repaired and checked by reading, not compiled, not run on hardware
//
#include <hip/hip_runtime.h>

typedef __attribute__((ext_vector_type(16))) _Float16 v16h;
typedef __attribute__((ext_vector_type(8)))  float    v8f;
typedef __attribute__((ext_vector_type(2)))  _Float16 h2;
typedef __attribute__((ext_vector_type(4)))  int      v4i;

typedef __attribute__((address_space(1))) v4i gv4i;  // global v4i
typedef __attribute__((address_space(3))) v4i lv4i;  // LDS v4i

#define N_NODES 4096
#define E_EDGES 4096
#define F_DIM   256
#define H_DIM   256

#if __has_builtin(__builtin_amdgcn_global_load_async_to_lds_b128)
#define USE_ASYNC_LDS 1
#else
#define USE_ASYNC_LDS 0
#endif

__device__ __forceinline__ void wait_async_loads() {
#if USE_ASYNC_LDS
#if __has_builtin(__builtin_amdgcn_s_wait_asynccnt)
    __builtin_amdgcn_s_wait_asynccnt(0);
#else
    asm volatile("s_wait_asynccnt 0x0" ::: "memory");
#endif
#endif
}

// ---------------------------------------------------------------------------
// Kernel 1: h = relu(Xf @ Ws + Xm @ Wh + bs + bh), output f16 row-major [N,H]
// One wave per 16x16 output tile. WMMA f32_16x16x32_f16, K-loop of 8 steps,
// two GEMMs accumulated into the same v8f accumulator.
// ---------------------------------------------------------------------------
__global__ __launch_bounds__(32)
void gemm_relu_f16(const float* __restrict__ posF, const float* __restrict__ posM,
                   const float* __restrict__ negF, const float* __restrict__ negM,
                   const float* __restrict__ Ws,   const float* __restrict__ Wh,
                   const float* __restrict__ bs,   const float* __restrict__ bh,
                   _Float16* __restrict__ Hout)
{
    const int lane = threadIdx.x;          // 0..31
    const int mt   = blockIdx.x;           // 0..255  (M tile)
    const int nt   = blockIdx.y;           // 0..15   (N tile)
    const int br   = blockIdx.z;           // 0..1    (branch)

    const float* A1 = br ? negF : posF;
    const float* A2 = br ? negM : posM;
    _Float16* Hb = Hout + (size_t)br * (N_NODES * H_DIM);

    const int m      = lane & 15;
    const int kband  = (lane >> 4) << 3;   // A: 0 or 8
    const int kcband = (lane >> 4) << 4;   // unused: 0 or 16
    const int col    = nt * 16 + (lane & 15);

    const float* arow1 = A1 + (size_t)(mt * 16 + m) * F_DIM;
    const float* arow2 = A2 + (size_t)(mt * 16 + m) * F_DIM;

    v8f acc = {};
    for (int k0 = 0; k0 < F_DIM; k0 += 32) {
        v16h a1, a2, b1, b2;
#pragma unroll
        for (int i = 0; i < 8; ++i) {
            // A fragment: 16-bit A 16x32 lane layout (half-wave K-bands of 8)
            const int ka = ((i >> 2) << 4) + kband + ((i & 3) << 1);
            float2 f1 = *(const float2*)(arow1 + k0 + ka);
            float2 f2 = *(const float2*)(arow2 + k0 + ka);
            a1[2 * i] = (_Float16)f1.x; a1[2 * i + 1] = (_Float16)f1.y;
            a2[2 * i] = (_Float16)f2.x; a2[2 * i + 1] = (_Float16)f2.y;
            // B fragment: 32x16, same K order as A (half-wave K-bands of 8),
            // two K values packed per VGPR
            const int kb = ((i >> 2) << 4) + kband + ((i & 3) << 1);
            b1[2 * i]     = (_Float16)Ws[(size_t)(k0 + kb)     * H_DIM + col];
            b1[2 * i + 1] = (_Float16)Ws[(size_t)(k0 + kb + 1) * H_DIM + col];
            b2[2 * i]     = (_Float16)Wh[(size_t)(k0 + kb)     * H_DIM + col];
            b2[2 * i + 1] = (_Float16)Wh[(size_t)(k0 + kb + 1) * H_DIM + col];
        }
        acc = __builtin_amdgcn_wmma_f32_16x16x32_f16(false, a1, false, b1,
                                                     (short)0, acc, false, false);
        acc = __builtin_amdgcn_wmma_f32_16x16x32_f16(false, a2, false, b2,
                                                     (short)0, acc, false, false);
    }

    const float bias   = bs[col] + bh[col];
    const int rowbase  = mt * 16 + ((lane >> 4) << 3); // C/D: lanes16-31 => M=r+8
#pragma unroll
    for (int r = 0; r < 8; ++r) {
        float v = fmaxf(acc[r] + bias, 0.0f);
        Hb[(size_t)(rowbase + r) * H_DIM + col] = (_Float16)v;
    }
}

// ---------------------------------------------------------------------------
// Kernel 2: masked max-min pooling (packed f16) fused with score head.
// Block = 256 threads = 8 waves = 8 hyperedges. 64-node tiles of h staged into
// double-buffered LDS (2 x 32 KB) via GLOBAL_LOAD_ASYNC_TO_LDS_B128 so the DMA
// overlaps the ballot-driven pk_max/pk_min inner loop.
// ---------------------------------------------------------------------------
__global__ __launch_bounds__(256)
void pool_score(const h2* __restrict__ Hall, const int* __restrict__ mask,
                const float* __restrict__ Wsc, const float* __restrict__ bsc,
                float* __restrict__ out)
{
    __shared__ __align__(16) h2 smem[2][64][H_DIM / 2];  // 2 x 32 KB

    const int lane = threadIdx.x & 31;
    const int wave = threadIdx.x >> 5;
    const int br   = blockIdx.y;
    const int e    = blockIdx.x * 8 + wave;

    const h2* Hb    = Hall + (size_t)br * (N_NODES * (H_DIM / 2));
    const int* mrow = mask + (size_t)e * N_NODES;

    // stage one 64-row x 512B tile into smem[buf]; each wave loads 8 rows,
    // one b128 per lane per row
    auto stage = [&](int nb, int buf) {
        for (int r = wave; r < 64; r += 8) {
            const uint4* src = (const uint4*)(Hb + (size_t)(nb + r) * (H_DIM / 2));
            uint4* dst = ((uint4*)&smem[buf][r][0]) + lane;
#if USE_ASYNC_LDS
            __builtin_amdgcn_global_load_async_to_lds_b128(
                (gv4i*)(src + lane), (lv4i*)dst, 0, 0);
#else
            *dst = src[lane];
#endif
        }
    };

    const _Float16 hneg = (_Float16)(-__builtin_huge_valf());
    const _Float16 hpos = (_Float16)( __builtin_huge_valf());
    h2 mx[4], mn[4];
#pragma unroll
    for (int i = 0; i < 4; ++i) {
        mx[i].x = hneg; mx[i].y = hneg;
        mn[i].x = hpos; mn[i].y = hpos;
    }

    stage(0, 0);

    int buf = 0;
    for (int nb = 0; nb < N_NODES; nb += 64, buf ^= 1) {
        wait_async_loads();     // my staging loads for smem[buf] are complete
        __syncthreads();        // everyone's loads done; prev compute done

        if (nb + 64 < N_NODES)  // prefetch next tile into the other buffer
            stage(nb + 64, buf ^ 1);

        // ballot the 64-node mask chunk into a 64-bit SGPR-resident bitmask
        unsigned m0 = (unsigned)__ballot(mrow[nb + lane] != 0);
        unsigned m1 = (unsigned)__ballot(mrow[nb + 32 + lane] != 0);
        unsigned long long bal = ((unsigned long long)m1 << 32) | (unsigned long long)m0;

        while (bal) {
            const int j = __builtin_ctzll(bal);
            bal &= bal - 1;
            const h2* row = &smem[buf][j][lane * 4];
            h2 v0 = row[0], v1 = row[1], v2 = row[2], v3 = row[3];
            mx[0] = __builtin_elementwise_max(mx[0], v0);
            mn[0] = __builtin_elementwise_min(mn[0], v0);
            mx[1] = __builtin_elementwise_max(mx[1], v1);
            mn[1] = __builtin_elementwise_min(mn[1], v1);
            mx[2] = __builtin_elementwise_max(mx[2], v2);
            mn[2] = __builtin_elementwise_min(mn[2], v2);
            mx[3] = __builtin_elementwise_max(mx[3], v3);
            mn[3] = __builtin_elementwise_min(mn[3], v3);
        }
    }

    // score head: (mx - mn) . W_score, wave reduction, sigmoid
    float sum = 0.0f;
#pragma unroll
    for (int i = 0; i < 4; ++i) {
        float2 w = *(const float2*)(Wsc + lane * 8 + 2 * i);
        float dx = (float)mx[i].x - (float)mn[i].x;
        float dy = (float)mx[i].y - (float)mn[i].y;
        sum += dx * w.x + dy * w.y;
    }
#pragma unroll
    for (int off = 16; off >= 1; off >>= 1)
        sum += __shfl_xor(sum, off, 32);

    if (lane == 0) {
        float s = sum + bsc[0];
        out[(size_t)br * E_EDGES + e] = 1.0f / (1.0f + __expf(-s));
    }
}

extern "C" void kernel_launch(void* const* d_in, const int* in_sizes, int n_in,
                              void* d_out, int out_size, void* d_ws, size_t ws_size,
                              hipStream_t stream) {
    const float* posF = (const float*)d_in[0];
    const float* posM = (const float*)d_in[1];
    const float* negF = (const float*)d_in[2];
    const float* negM = (const float*)d_in[3];
    const int*   mask = (const int*)  d_in[4];
    const float* Ws   = (const float*)d_in[5];
    const float* bs   = (const float*)d_in[6];
    const float* Wh   = (const float*)d_in[7];
    const float* bh   = (const float*)d_in[8];
    const float* Wsc  = (const float*)d_in[9];
    const float* bsc  = (const float*)d_in[10];

    _Float16* h = (_Float16*)d_ws;   // 2 branches x 4096 x 256 f16 = 4 MB

    dim3 gg(N_NODES / 16, H_DIM / 16, 2);
    gemm_relu_f16<<<gg, 32, 0, stream>>>(posF, posM, negF, negM, Ws, Wh, bs, bh, h);

    dim3 gp(E_EDGES / 8, 2);
    pool_score<<<gp, 256, 0, stream>>>((const h2*)h, mask, Wsc, bsc,
                                       (float*)d_out);
}
